// MultiHeadAttentionLayer_62216896249917
// MI455X (gfx1250) — compile-verified, matrix operand order checked by reading, not run on hardware
//
#include <hip/hip_runtime.h>

typedef __attribute__((ext_vector_type(16))) __bf16 v16bf;
typedef __attribute__((ext_vector_type(8)))  float  v8f;
typedef __attribute__((ext_vector_type(4)))  __bf16 v4bf;
typedef int b128_t __attribute__((vector_size(16)));   // matches builtin param

#define DEV static __device__ __forceinline__

constexpr int Bz = 2, Sq = 2048, Dm = 1024, Hn = 16;   // HD = 64
constexpr int Mtot = Bz * Sq;                          // 4096 rows

// ---------- CDNA5 async global->LDS copy (ASYNCcnt) --------------------------

#if __has_builtin(__builtin_amdgcn_global_load_async_to_lds_b128) && \
    __has_builtin(__builtin_amdgcn_s_wait_asynccnt)
#define USE_ASYNC 1
#define WAIT_ASYNC(n) __builtin_amdgcn_s_wait_asynccnt(n)
#else
#define USE_ASYNC 0
#define WAIT_ASYNC(n)
#endif

// copy 16 bytes (8 bf16) global -> LDS
DEV void copy16(const __bf16* g, __bf16* l) {
#if USE_ASYNC
  __builtin_amdgcn_global_load_async_to_lds_b128(
      (__attribute__((address_space(1))) b128_t*)(g),
      (__attribute__((address_space(3))) b128_t*)(l), 0, 0);
#else
  *(uint4*)l = *(const uint4*)g;
#endif
}

// ---------- helpers ----------------------------------------------------------

DEV __bf16 f2bf(float f) {                 // round-to-nearest-even f32 -> bf16
  unsigned u = __builtin_bit_cast(unsigned, f);
  unsigned r = (u + 0x7FFFu + ((u >> 16) & 1u)) >> 16;
  unsigned short s = (unsigned short)r;
  return __builtin_bit_cast(__bf16, s);
}

// Load a 16-bit A/B-style WMMA fragment (16x32, K-major rows).
// Caller passes &row[lane_chunk_base]; layout per CDNA5 ISA 7.12.2:
// lanes 0-15: K c0..c0+7 (q[0]) and c0+16..c0+23 (q[1]); lanes 16-31: +8.
DEV v16bf load_frag(const __bf16* p) {
  union { v16bf v; uint4 q[2]; } u;
  u.q[0] = *(const uint4*)(p);
  u.q[1] = *(const uint4*)(p + 16);
  return u.v;
}

DEV v8f wmma_bf16(v16bf a, v16bf b, v8f c) {
  return __builtin_amdgcn_wmma_f32_16x16x32_bf16(false, a, false, b,
                                                 (short)0, c, false, false);
}

// ---------- fp32 -> bf16 conversion (x) --------------------------------------

__global__ __launch_bounds__(256)
void f32_to_bf16_vec(const float* __restrict__ in, __bf16* __restrict__ out,
                     int n4) {
  int i = blockIdx.x * blockDim.x + threadIdx.x;
  if (i < n4) {
    float4 f = ((const float4*)in)[i];
    v4bf r;
    r[0] = f2bf(f.x); r[1] = f2bf(f.y); r[2] = f2bf(f.z); r[3] = f2bf(f.w);
    ((v4bf*)out)[i] = r;
  }
}

// ---------- one-time weight transpose: W(KxN f32) -> Wt(NxK bf16) ------------

__global__ __launch_bounds__(256)
void transpose_f32_to_bf16(const float* __restrict__ in,
                           __bf16* __restrict__ out) {
  __shared__ float tile[32][33];
  const int nb = blockIdx.x * 32, kb = blockIdx.y * 32;
  const int tx = threadIdx.x & 31, ty = threadIdx.x >> 5;   // 32 x 8
#pragma unroll
  for (int i = ty; i < 32; i += 8)
    tile[i][tx] = in[(size_t)(kb + i) * Dm + nb + tx];      // coalesced read
  __syncthreads();
#pragma unroll
  for (int i = ty; i < 32; i += 8)
    out[(size_t)(nb + i) * Dm + kb + tx] = f2bf(tile[tx][i]); // coalesced write
}

// ---------- GEMM:  out(MxN) = A(MxK,bf16) @ Wt^T + bias ----------------------
// A row-major [M][K]; Wt is the N x K (pre-transposed) bf16 weight.
// BM=BN=128, BK=64; 256 threads = 8 waves (2x4); wave tile 64x32 = 8 WMMA accs.
// Double-buffered LDS fed by async global->LDS copies.
// OUT_MODE: 0 = bf16 row-major, 1 = bf16 transposed [N][M], 2 = f32 row-major.

template <int OUT_MODE>
__global__ __launch_bounds__(256)
void gemm_bias_kernel(const __bf16* __restrict__ A,
                      const __bf16* __restrict__ Wt,
                      const float* __restrict__ bias, void* __restrict__ outv) {
  constexpr int Kd = Dm, Nd = Dm;
  __shared__ __align__(16) __bf16 sA[2][128][72];   // [m][k]
  __shared__ __align__(16) __bf16 sB[2][128][72];   // [n][k]

  const int tid = threadIdx.x, lane = tid & 31;
  const int l15 = lane & 15, lh = lane >> 4;
  const int wv = tid >> 5, wm = wv & 1, wn = wv >> 1;
  const int mbase = blockIdx.y * 128, nbase = blockIdx.x * 128;

  // staging assignment: thread covers 64 bytes of one row in each tile
  const int sr = tid >> 1, scb = (tid & 1) * 32;

  v8f acc[4][2];
#pragma unroll
  for (int i = 0; i < 4; ++i)
#pragma unroll
    for (int j = 0; j < 2; ++j)
#pragma unroll
      for (int v = 0; v < 8; ++v) acc[i][j][v] = 0.0f;

  auto issue = [&](int buf, int kb) {
    const __bf16* ga = A  + (size_t)(mbase + sr) * Kd + kb + scb;
    const __bf16* gb = Wt + (size_t)(nbase + sr) * Kd + kb + scb;
#pragma unroll
    for (int i = 0; i < 4; ++i) {
      copy16(ga + i * 8, &sA[buf][sr][scb + i * 8]);
      copy16(gb + i * 8, &sB[buf][sr][scb + i * 8]);
    }
  };

  int buf = 0;
  issue(0, 0);
  for (int kb = 0; kb < Kd; kb += 64) {
    if (kb + 64 < Kd) { issue(buf ^ 1, kb + 64); WAIT_ASYNC(8); }
    else              { WAIT_ASYNC(0); }
    __syncthreads();

#pragma unroll
    for (int ks = 0; ks < 2; ++ks) {
      const int c0 = lh * 8 + ks * 32;
      v16bf af[4], bfr[2];
#pragma unroll
      for (int i = 0; i < 4; ++i)
        af[i] = load_frag(&sA[buf][wm * 64 + i * 16 + l15][c0]);
#pragma unroll
      for (int j = 0; j < 2; ++j)
        bfr[j] = load_frag(&sB[buf][wn * 32 + j * 16 + l15][c0]);
#pragma unroll
      for (int i = 0; i < 4; ++i)
#pragma unroll
        for (int j = 0; j < 2; ++j)
          acc[i][j] = wmma_bf16(af[i], bfr[j], acc[i][j]);
    }
    __syncthreads();    // protect buffer we are about to overwrite next iter
    buf ^= 1;
  }

  // epilogue: + bias, store (C layout: lane=col, vgpr v = row v + 8*lh)
#pragma unroll
  for (int j = 0; j < 2; ++j) {
    const int col = nbase + wn * 32 + j * 16 + l15;
    const float bb = bias[col];
#pragma unroll
    for (int i = 0; i < 4; ++i) {
      const int row0 = mbase + wm * 64 + i * 16 + 8 * lh;
#pragma unroll
      for (int v = 0; v < 8; ++v) {
        float val = acc[i][j][v] + bb;
        if (OUT_MODE == 2)
          ((float*)outv)[(size_t)(row0 + v) * Nd + col] = val;
        else if (OUT_MODE == 1)   // transposed bf16: out[col][row]
          ((__bf16*)outv)[(size_t)col * Mtot + (row0 + v)] = f2bf(val);
        else
          ((__bf16*)outv)[(size_t)(row0 + v) * Nd + col] = f2bf(val);
      }
    }
  }
}

// ---------- causal flash attention (per (b,h), 128 query rows / block) ------
// 8 waves; wave w owns query rows [qbase+16w, qbase+16w+15]; K blocks of 64.
// K staged row-major (== NxK B-fragment layout for QK^T); V staged from the
// pre-transposed global Vt [D][B*S] so P@V B-fragments are contiguous copies.

__global__ __launch_bounds__(256)
void attn_kernel(const __bf16* __restrict__ Q, const __bf16* __restrict__ K,
                 const __bf16* __restrict__ Vt, __bf16* __restrict__ O) {
  __shared__ __align__(16) __bf16 sK[2][64][72];     // [keyrow][hd]
  __shared__ __align__(16) __bf16 sV[2][64][72];     // [hd][keyrow]
  __shared__ __align__(16) __bf16 sP[8][16][72];     // per-wave P tile

  const int qbase = blockIdx.x * 128;
  const int h = blockIdx.y, b = blockIdx.z;
  const int tid = threadIdx.x, lane = tid & 31, w = tid >> 5;
  const int l15 = lane & 15, lh = lane >> 4;
  const size_t headoff = (size_t)h * 64;

  // staging assignment: thread covers 32 bytes of one row in each tile
  const int sr = tid >> 2, scb = (tid & 3) * 16;

  // Q fragments (A layout), loaded straight from global
  const int qrowA = qbase + w * 16 + l15;
  const __bf16* qp = Q + ((size_t)b * Sq + qrowA) * Dm + headoff;
  v16bf aq[2];
#pragma unroll
  for (int ks = 0; ks < 2; ++ks) aq[ks] = load_frag(qp + lh * 8 + ks * 32);

  v8f o4[4];
  float mrow[8], lrow[8];
#pragma unroll
  for (int t = 0; t < 4; ++t)
#pragma unroll
    for (int v = 0; v < 8; ++v) o4[t][v] = 0.0f;
#pragma unroll
  for (int v = 0; v < 8; ++v) { mrow[v] = -1e30f; lrow[v] = 0.0f; }

  const int qr0 = qbase + w * 16 + 8 * lh;   // row of acc element v is qr0+v
  const int kend = qbase + 128;              // causal: no keys beyond block

  auto issue = [&](int buf, int kbb) {
    const __bf16* gk = K  + ((size_t)b * Sq + kbb + sr) * Dm + headoff + scb;
    const __bf16* gv = Vt + (headoff + sr) * Mtot + (size_t)b * Sq + kbb + scb;
    copy16(gk,     &sK[buf][sr][scb]);
    copy16(gk + 8, &sK[buf][sr][scb + 8]);
    copy16(gv,     &sV[buf][sr][scb]);
    copy16(gv + 8, &sV[buf][sr][scb + 8]);
  };

  int buf = 0;
  issue(0, 0);
  for (int kbb = 0; kbb < kend; kbb += 64) {
    if (kbb + 64 < kend) { issue(buf ^ 1, kbb + 64); WAIT_ASYNC(4); }
    else                 { WAIT_ASYNC(0); }
    __syncthreads();

    // S = Q K^T : 16x64 per wave (4 N tiles x 2 K steps)
    v8f sc[4];
#pragma unroll
    for (int t = 0; t < 4; ++t)
#pragma unroll
      for (int v = 0; v < 8; ++v) sc[t][v] = 0.0f;
#pragma unroll
    for (int ks = 0; ks < 2; ++ks) {
      const int c0 = lh * 8 + ks * 32;
#pragma unroll
      for (int t = 0; t < 4; ++t) {
        v16bf bk = load_frag(&sK[buf][t * 16 + l15][c0]);
        sc[t] = wmma_bf16(aq[ks], bk, sc[t]);
      }
    }

    // scale + causal mask
#pragma unroll
    for (int t = 0; t < 4; ++t) {
      const int kc = kbb + t * 16 + l15;
#pragma unroll
      for (int v = 0; v < 8; ++v) {
        float x = sc[t][v] * 0.125f;               // 1/sqrt(64)
        sc[t][v] = (kc <= qr0 + v) ? x : -1e30f;
      }
    }

    // online softmax; rows live across the 16 lanes of each half
#pragma unroll
    for (int v = 0; v < 8; ++v) {
      float mx = sc[0][v];
#pragma unroll
      for (int t = 1; t < 4; ++t) mx = fmaxf(mx, sc[t][v]);
#pragma unroll
      for (int off = 1; off < 16; off <<= 1)
        mx = fmaxf(mx, __shfl_xor(mx, off, 32));
      const float mnew = fmaxf(mrow[v], mx);
      const float corr = __expf(mrow[v] - mnew);
      float rs = 0.0f;
#pragma unroll
      for (int t = 0; t < 4; ++t) {
        float p = __expf(sc[t][v] - mnew);
        sc[t][v] = p;
        rs += p;
      }
#pragma unroll
      for (int off = 1; off < 16; off <<= 1) rs += __shfl_xor(rs, off, 32);
      lrow[v] = lrow[v] * corr + rs;
      mrow[v] = mnew;
#pragma unroll
      for (int t = 0; t < 4; ++t) o4[t][v] *= corr;
    }

    // C-layout -> A-layout via wave-private LDS (DS ops in-order per wave)
#pragma unroll
    for (int t = 0; t < 4; ++t)
#pragma unroll
      for (int v = 0; v < 8; ++v)
        sP[w][v + 8 * lh][t * 16 + l15] = f2bf(sc[t][v]);

    // O += P @ V  (4 HD tiles x 2 K steps)
#pragma unroll
    for (int ks = 0; ks < 2; ++ks) {
      const int c0 = lh * 8 + ks * 32;
      v16bf ap = load_frag(&sP[w][l15][c0]);
#pragma unroll
      for (int t = 0; t < 4; ++t) {
        v16bf bv = load_frag(&sV[buf][t * 16 + l15][c0]);
        o4[t] = wmma_bf16(ap, bv, o4[t]);
      }
    }
    __syncthreads();    // protect buffer overwritten by next iteration's issue
    buf ^= 1;
  }

  // normalize and store as bf16 into (B,S,D) layout
#pragma unroll
  for (int v = 0; v < 8; ++v) {
    const float inv_l = 1.0f / lrow[v];
    const int qr = qr0 + v;
#pragma unroll
    for (int t = 0; t < 4; ++t)
      O[((size_t)b * Sq + qr) * Dm + headoff + t * 16 + l15] =
          f2bf(o4[t][v] * inv_l);
  }
}

// ---------- launcher ---------------------------------------------------------

extern "C" void kernel_launch(void* const* d_in, const int* in_sizes, int n_in,
                              void* d_out, int out_size, void* d_ws,
                              size_t ws_size, hipStream_t stream) {
  (void)in_sizes; (void)n_in; (void)out_size; (void)ws_size;
  const float* x  = (const float*)d_in[0];
  const float* Wq = (const float*)d_in[1];
  const float* bq = (const float*)d_in[2];
  const float* Wk = (const float*)d_in[3];
  const float* bk = (const float*)d_in[4];
  const float* Wv = (const float*)d_in[5];
  const float* bv = (const float*)d_in[6];
  const float* Wo = (const float*)d_in[7];
  const float* bo = (const float*)d_in[8];
  // d_in[9] = causal mask: computed analytically in-kernel, not read.
  float* out = (float*)d_out;

  char* ws = (char*)d_ws;
  const size_t elems = (size_t)Mtot * Dm;     // 4M elements per (B,S,D) tensor
  const size_t welem = (size_t)Dm * Dm;       // 1M elements per weight
  __bf16* xb  = (__bf16*)(ws);                               // 8 MB
  __bf16* Qb  = (__bf16*)(ws + elems * 2);                   // 8 MB
  __bf16* Kb  = (__bf16*)(ws + elems * 4);                   // 8 MB
  __bf16* Vtb = (__bf16*)(ws + elems * 6);                   // 8 MB  [D][B*S]
  __bf16* Ab  = (__bf16*)(ws + elems * 8);                   // 8 MB
  __bf16* Wqt = (__bf16*)(ws + elems * 10);                  // 2 MB each
  __bf16* Wkt = Wqt + welem;
  __bf16* Wvt = Wkt + welem;
  __bf16* Wot = Wvt + welem;                                 // 48 MB total

  const int n4 = (int)(elems / 4);
  f32_to_bf16_vec<<<dim3((n4 + 255) / 256), dim3(256), 0, stream>>>(x, xb, n4);

  const dim3 tgrid(Dm / 32, Dm / 32);
  transpose_f32_to_bf16<<<tgrid, 256, 0, stream>>>(Wq, Wqt);
  transpose_f32_to_bf16<<<tgrid, 256, 0, stream>>>(Wk, Wkt);
  transpose_f32_to_bf16<<<tgrid, 256, 0, stream>>>(Wv, Wvt);
  transpose_f32_to_bf16<<<tgrid, 256, 0, stream>>>(Wo, Wot);

  const dim3 ggrid(Dm / 128, Mtot / 128);     // (8, 32)
  gemm_bias_kernel<0><<<ggrid, 256, 0, stream>>>(xb, Wqt, bq, (void*)Qb);
  gemm_bias_kernel<0><<<ggrid, 256, 0, stream>>>(xb, Wkt, bk, (void*)Kb);
  gemm_bias_kernel<1><<<ggrid, 256, 0, stream>>>(xb, Wvt, bv, (void*)Vtb);

  attn_kernel<<<dim3(Sq / 128, Hn, Bz), dim3(256), 0, stream>>>(Qb, Kb, Vtb,
                                                                Ab);

  gemm_bias_kernel<2><<<ggrid, 256, 0, stream>>>(Ab, Wot, bo, (void*)out);
}
